// TGNModel_6648609374720
// MI455X (gfx1250) — matrix operand order repaired and checked by reading, not compiled, not run on hardware
//
#include <hip/hip_runtime.h>

typedef __attribute__((ext_vector_type(16))) _Float16 v16h;
typedef __attribute__((ext_vector_type(8)))  _Float16 v8h;
typedef __attribute__((ext_vector_type(8)))  float    v8f;

constexpr int MEM_DIM  = 100;
constexpr int TIME_DIM = 100;
constexpr int EDGE_DIM = 172;
constexpr int HIDDEN   = 128;
constexpr int IN_DIM   = 472;          // 2*MEM + TIME + EDGE
constexpr int K_PAD    = 480;          // 15 chunks of 32
constexpr int NKK      = 15;           // K chunks
constexpr int ROW_PAD  = 488;          // halves per Xs row (976B, 16B-aligned stride)
constexpr int TILE     = 128;          // edges per workgroup (8 waves x 16 rows)

__global__ __launch_bounds__(256, 1)
void tgn_growth_kernel(const int* __restrict__ src, const int* __restrict__ dst,
                       const float* __restrict__ t, const float* __restrict__ edge_attr,
                       const float* __restrict__ memory, const float* __restrict__ last_update,
                       const float* __restrict__ time_w, const float* __restrict__ time_b,
                       const float* __restrict__ w1, const float* __restrict__ b1,
                       const float* __restrict__ w2, const float* __restrict__ b2,
                       float* __restrict__ out, int E)
{
    // 128*488*2 = 124,928 B  +  15*8*32*16*2 = 122,880 B  => 247,808 B LDS (<320KB/WGP)
    __shared__ _Float16 Xs[TILE][ROW_PAD];
    __shared__ _Float16 Ws[NKK][8][32][16];   // [kchunk][ntile][lane][half] B-fragment swizzle

    const int tid  = threadIdx.x;
    const int lane = tid & 31;
    const int wave = tid >> 5;
    const int base = blockIdx.x * TILE;

    // ---------------- Stage w1 -> LDS in WMMA B-fragment layout (f32 -> f16) -------------
    // Flat over (k, n), n fastest => coalesced reads of L2-resident w1.
    // B 32x16 f16 layout used here (contiguous k, not AMD's table order): lane = (k%32 >=16 ? 16:0) + (n%16); half index h = k%16.
    for (int idx = tid; idx < K_PAD * HIDDEN; idx += 256) {
        int k = idx >> 7;            // 0..479
        int n = idx & 127;           // 0..127
        float v = (k < IN_DIM) ? w1[k * HIDDEN + n] : 0.0f;
        int kk  = k >> 5;
        int rem = k & 31;
        int l   = ((rem >> 4) << 4) | (n & 15);
        int h   = rem & 15;
        int j   = n >> 4;
        Ws[kk][j][l][h] = (_Float16)v;
    }

    // ---------------- Gather + time-encode + concat into Xs (f16) ------------------------
    for (int rr = 0; rr < TILE / 8; ++rr) {          // 16 rows per wave
        int row = wave * 16 + rr;
        int e   = base + row;
        bool valid = (e < E);
        int s = 0, d = 0;
        float ti = 0.0f, lu = 0.0f;
        if (valid) { s = src[e]; d = dst[e]; ti = t[e]; lu = last_update[s]; }
        float dtv = ti - lu;
        for (int c = lane; c < ROW_PAD; c += 32) {
            float v = 0.0f;
            if (valid) {
                if (c < MEM_DIM) {
                    v = memory[(long)s * MEM_DIM + c];
                } else if (c < 2 * MEM_DIM) {
                    v = memory[(long)d * MEM_DIM + (c - MEM_DIM)];
                } else if (c < 2 * MEM_DIM + TIME_DIM) {
                    int q = c - 2 * MEM_DIM;
                    v = __cosf(dtv * time_w[q] + time_b[q]);
                } else if (c < IN_DIM) {
                    v = edge_attr[(long)e * EDGE_DIM + (c - (2 * MEM_DIM + TIME_DIM))];
                }
            }
            Xs[row][c] = (_Float16)v;
        }
    }
    __syncthreads();

    // ---------------- per-lane b1 / w2 slices (n = j*16 + lane%16) ------------------------
    float b1v[8], w2v[8];
#pragma unroll
    for (int j = 0; j < 8; ++j) {
        int n = j * 16 + (lane & 15);
        b1v[j] = b1[n];
        w2v[j] = w2[n];
    }
    const float b2v = b2[0];

    // ---------------- WMMA GEMM: [16 x 472] @ [472 x 128] --------------------------------
    v8f acc[8];
    {
        v8f z = {};
#pragma unroll
        for (int j = 0; j < 8; ++j) acc[j] = z;
    }

    // A fragment (16-bit A 16x32): lanes 0-15 -> K 0..15; lanes 16-31 -> K 16..31 (matches Ws B layout)
    const _Float16* xrow = &Xs[wave * 16 + (lane & 15)][0];
    const int khalf = (lane >> 4) * 16;

    for (int kk = 0; kk < NKK; ++kk) {
        union { v16h v; struct { v8h lo, hi; } p; } a;
        a.p.lo = *(const v8h*)(xrow + kk * 32 + khalf);
        a.p.hi = *(const v8h*)(xrow + kk * 32 + khalf + 8);
#pragma unroll
        for (int j = 0; j < 8; ++j) {
            union { v16h v; struct { v8h lo, hi; } p; } b;
            const _Float16* wp = &Ws[kk][j][lane][0];
            b.p.lo = *(const v8h*)(wp);
            b.p.hi = *(const v8h*)(wp + 8);
            acc[j] = __builtin_amdgcn_wmma_f32_16x16x32_f16(
                false, a.v, false, b.v, (short)0, acc[j], false, false);
        }
    }

    // ---------------- Fused epilogue: relu(h + b1) . w2  + b2 -----------------------------
    // C layout: acc[j][r] = h[M][j*16 + lane%16], M = r + 8*(lane>=16)
    float part[8];
#pragma unroll
    for (int r = 0; r < 8; ++r) {
        float sum = 0.0f;
#pragma unroll
        for (int j = 0; j < 8; ++j) {
            float h = acc[j][r] + b1v[j];
            h = fmaxf(h, 0.0f);
            sum = fmaf(h, w2v[j], sum);
        }
        part[r] = sum;
    }
    // Reduce over the 16 lanes of each half-group (xor 1,2,4,8 stays within the half)
#pragma unroll
    for (int r = 0; r < 8; ++r) {
        float s = part[r];
        s += __shfl_xor(s, 1);
        s += __shfl_xor(s, 2);
        s += __shfl_xor(s, 4);
        s += __shfl_xor(s, 8);
        part[r] = s;
    }
    if ((lane & 15) == 0) {
        int mbase = base + wave * 16 + ((lane >> 4) << 3);   // lanes<16: rows 0..7, lanes>=16: rows 8..15
#pragma unroll
        for (int r = 0; r < 8; ++r) {
            int e = mbase + r;
            if (e < E) out[e] = part[r] + b2v;
        }
    }
}

extern "C" void kernel_launch(void* const* d_in, const int* in_sizes, int n_in,
                              void* d_out, int out_size, void* d_ws, size_t ws_size,
                              hipStream_t stream) {
    const int*   src         = (const int*)  d_in[0];
    const int*   dst         = (const int*)  d_in[1];
    const float* t           = (const float*)d_in[2];
    const float* edge_attr   = (const float*)d_in[3];
    const float* memory      = (const float*)d_in[4];
    const float* last_update = (const float*)d_in[5];
    const float* time_w      = (const float*)d_in[6];
    const float* time_b      = (const float*)d_in[7];
    const float* w1          = (const float*)d_in[8];
    const float* b1          = (const float*)d_in[9];
    const float* w2          = (const float*)d_in[10];
    const float* b2          = (const float*)d_in[11];
    float* out = (float*)d_out;

    const int E = in_sizes[0];
    const int grid = (E + TILE - 1) / TILE;
    tgn_growth_kernel<<<grid, 256, 0, stream>>>(src, dst, t, edge_attr, memory, last_update,
                                                time_w, time_b, w1, b1, w2, b2, out, E);
}
